// TimeLSTM_66855460929587
// MI455X (gfx1250) — matrix operand order repaired and checked by reading, not compiled, not run on hardware
//
#include <hip/hip_runtime.h>

// ---------------------------------------------------------------------------
// TimeLSTM fused scan for gfx1250 (MI455X), wave32 + v_wmma_f32_16x16x32_bf16
//   B=256, S=2048, E=64, H=128, O=1
//   16 blocks (one per 16-row batch tile) x 640 threads (20 waves):
//     waves 0..15 : gate GEMM tiles (x@U.T + h@W_all.T), sigmoid -> LDS
//     waves 16..19: c@W_d.T (tanh) + per-step state update (c,h fp32 in regs)
//   Roles are split into two disjoint scan loops (same barrier schedule) so
//   the register allocator overlays their live ranges -> no scratch spills.
//   A-fragments are batch-preloaded per step so LDS latency is paid once,
//   and global loads (dt, next x) are hoisted off the update critical path.
// ---------------------------------------------------------------------------

typedef __attribute__((ext_vector_type(16))) __bf16 v16bf;
typedef __attribute__((ext_vector_type(8)))  __bf16 v8bf;
typedef __attribute__((ext_vector_type(8)))  float  v8f;

#define NB 256
#define NS 2048
#define NE 64
#define NH 128
#define BT 16
#define WAVES_GATE 16
#define WAVES_WD   4
#define NTHREADS   ((WAVES_GATE + WAVES_WD) * 32)   // 640

#define HA_ST 136   // bf16 elems/row for h,c tiles (128 + 8 pad -> conflict-free b128 reads)
#define XA_ST 72    // bf16 elems/row for x tile (64 + 8 pad)
#define G_ST  516   // f32 elems/row for gate buffer (512 + 4 pad -> conflict-free)

#define LOG2E 1.4426950408889634f

__device__ __forceinline__ float fexp2(float x){ return __builtin_amdgcn_exp2f(x); }
__device__ __forceinline__ float frcp (float x){ return __builtin_amdgcn_rcpf(x); }
__device__ __forceinline__ float sigmoidf_(float x){ return frcp(1.0f + fexp2(-LOG2E * x)); }
__device__ __forceinline__ float tanhf_(float x){ return 1.0f - 2.0f * frcp(fexp2(2.0f * LOG2E * x) + 1.0f); }
__device__ __forceinline__ __bf16 f2bf(float x){ return (__bf16)x; }

// A-fragment (16x32 bf16, MxK) from row-major LDS [16][stride]:
//   lane<16 : row=lane,    elems 0..7 = K kc*32+0..7,  elems 8..15 = K kc*32+16..23
//   lane>=16: row=lane-16, elems 0..7 = K kc*32+8..15, elems 8..15 = K kc*32+24..31
__device__ __forceinline__ v16bf afrag_lds(const __bf16* base, int stride, int kc, int lane){
  const int r  = lane & 15;
  const int k0 = kc * 32 + ((lane & 16) ? 8 : 0);
  const __bf16* p = base + r * stride + k0;
  v8bf lo = *(const v8bf*)(p);
  v8bf hi = *(const v8bf*)(p + 16);
  v16bf a;
#pragma unroll
  for (int i = 0; i < 8; ++i){ a[i] = lo[i]; a[i + 8] = hi[i]; }
  return a;
}

// B-fragment (32x16 bf16, KxN) for out = A @ W^T (B[k][n] = W[n][k]),
// converted on the fly from row-major fp32 weights in global memory.
//   lane l: column n = n0 + (l&15); elems 0..7 = k kb+h*8+0..7, elems 8..15 = k kb+16+h*8+0..7 (h = l>=16)
__device__ __forceinline__ v16bf bfrag_gl(const float* M, int ld, int n0, int kb, int lane){
  const float* p = M + (size_t)(n0 + (lane & 15)) * ld + kb + ((lane & 16) ? 8 : 0);
  v16bf b;
#pragma unroll
  for (int i = 0; i < 16; ++i) b[i] = f2bf(p[i < 8 ? i : i + 8]);
  return b;
}

__device__ __forceinline__ v8f wmma_bf16(v16bf a, v16bf b, v8f c){
  return __builtin_amdgcn_wmma_f32_16x16x32_bf16(false, a, false, b, (short)0, c, false, false);
}

__global__ __launch_bounds__(NTHREADS, 1)
void timelstm_scan(const float* __restrict__ inputs,   // [B,S,E]
                   const float* __restrict__ tintv,    // [B,S]
                   const float* __restrict__ W_all,    // [4H,H]
                   const float* __restrict__ b_all,    // [4H]
                   const float* __restrict__ U_all,    // [4H,E]
                   const float* __restrict__ b_u,      // [4H]
                   const float* __restrict__ W_d,  const float* __restrict__ b_d,   // [H,H],[H]
                   const float* __restrict__ W_out,const float* __restrict__ b_out, // [1,H],[1]
                   float* __restrict__ out)            // [B]
{
  __shared__ __bf16 hA[16 * HA_ST];     // h state, bf16, A-layout source
  __shared__ __bf16 cA[16 * HA_ST];     // c state, bf16, A-layout source
  __shared__ __bf16 xA[16 * XA_ST];     // current-step input slice, bf16
  __shared__ float  gbuf[16 * G_ST];    // post-sigmoid gate preactivations
  __shared__ float  red[16];            // final output reduction

  const int tid  = threadIdx.x;
  const int wave = tid >> 5;
  const int lane = tid & 31;
  const int b0   = blockIdx.x * BT;
  const bool isGate = (wave < WAVES_GATE);

  // ---------------- init LDS ----------------
  for (int i = tid; i < 16 * HA_ST; i += NTHREADS){ hA[i] = f2bf(0.f); cA[i] = f2bf(0.f); }
  if (tid < 16) red[tid] = 0.f;
  if (!isGate){  // preload x tile for step 0 (4 wd-waves cover 16x64 elems)
    const int wd = wave - WAVES_GATE;
#pragma unroll
    for (int i = 0; i < 8; ++i){
      int idx = wd * 256 + i * 32 + lane;
      int row = idx >> 6, k = idx & 63;
      xA[row * XA_ST + k] = f2bf(inputs[((size_t)(b0 + row) * NS + 0) * NE + k]);
    }
  }
  __syncthreads();

  if (isGate){
    // =====================================================================
    // GATE WAVES: per step compute sigmoid(x@U.T + h@W_all.T + b) -> gbuf
    // =====================================================================
    const int n0 = wave * 32;                       // gate-column base (2 tiles)
    v16bf bw[2][4], bu[2][2];
#pragma unroll
    for (int t = 0; t < 2; ++t){
#pragma unroll
      for (int kc = 0; kc < 4; ++kc) bw[t][kc] = bfrag_gl(W_all, NH, n0 + 16 * t, kc * 32, lane);
#pragma unroll
      for (int kc = 0; kc < 2; ++kc) bu[t][kc] = bfrag_gl(U_all, NE, n0 + 16 * t, kc * 32, lane);
    }
    const float bias0 = b_all[n0      + (lane & 15)] + b_u[n0      + (lane & 15)];
    const float bias1 = b_all[n0 + 16 + (lane & 15)] + b_u[n0 + 16 + (lane & 15)];
    const int roff = (lane & 16) ? 8 : 0;
    const int c0   = n0 + (lane & 15);

    for (int s = 0; s < NS; ++s){
      // batch-preload all A fragments for this step (one dscnt wait total)
      v16bf ax[2], ah[4];
#pragma unroll
      for (int kc = 0; kc < 2; ++kc) ax[kc] = afrag_lds(xA, XA_ST, kc, lane);
#pragma unroll
      for (int kc = 0; kc < 4; ++kc) ah[kc] = afrag_lds(hA, HA_ST, kc, lane);

      v8f acc0, acc1;
#pragma unroll
      for (int i = 0; i < 8; ++i){ acc0[i] = bias0; acc1[i] = bias1; }
#pragma unroll
      for (int kc = 0; kc < 2; ++kc){                 // x @ U_all.T (K=64)
        acc0 = wmma_bf16(ax[kc], bu[0][kc], acc0);
        acc1 = wmma_bf16(ax[kc], bu[1][kc], acc1);
      }
#pragma unroll
      for (int kc = 0; kc < 4; ++kc){                 // h @ W_all.T (K=128)
        acc0 = wmma_bf16(ah[kc], bw[0][kc], acc0);
        acc1 = wmma_bf16(ah[kc], bw[1][kc], acc1);
      }
#pragma unroll
      for (int v = 0; v < 8; ++v){
        int row = v + roff;
        gbuf[row * G_ST + c0]      = sigmoidf_(acc0[v]);
        gbuf[row * G_ST + c0 + 16] = sigmoidf_(acc1[v]);
      }
      __syncthreads();   // gates visible to update waves
      __syncthreads();   // wait for h/c/x update before next step
    }
  } else {
    // =====================================================================
    // WD WAVES: per step compute c_s1 = tanh(c@W_d.T + b_d), then update
    //           c,h (fp32 in registers), publish bf16 h/c and next x tile
    // =====================================================================
    const int wd = wave - WAVES_GATE;
    const int n0 = wd * 32;                          // h-column base (2 tiles)
    v16bf bw[2][4];
#pragma unroll
    for (int t = 0; t < 2; ++t)
#pragma unroll
      for (int kc = 0; kc < 4; ++kc) bw[t][kc] = bfrag_gl(W_d, NH, n0 + 16 * t, kc * 32, lane);
    const float bias0 = b_d[n0      + (lane & 15)];
    const float bias1 = b_d[n0 + 16 + (lane & 15)];
    const int roff = (lane & 16) ? 8 : 0;
    const int cl   = lane & 15;

    float cstate[2][8], hsum[2][8];
#pragma unroll
    for (int t = 0; t < 2; ++t)
#pragma unroll
      for (int v = 0; v < 8; ++v){ cstate[t][v] = 0.f; hsum[t][v] = 0.f; }

    for (int s = 0; s < NS; ++s){
      // ---- issue global loads early: dt for step s, x for step s+1 ----
      float dval[8];
#pragma unroll
      for (int v = 0; v < 8; ++v)
        dval[v] = tintv[(size_t)(b0 + v + roff) * NS + s];
      float xv[8];
      if (s + 1 < NS){
#pragma unroll
        for (int i = 0; i < 8; ++i){
          int idx = wd * 256 + i * 32 + lane;
          xv[i] = inputs[((size_t)(b0 + (idx >> 6)) * NS + (s + 1)) * NE + (idx & 63)];
        }
      }

      // ---- c @ W_d.T (K=128), A-frags batch-preloaded ----
      v16bf ac[4];
#pragma unroll
      for (int kc = 0; kc < 4; ++kc) ac[kc] = afrag_lds(cA, HA_ST, kc, lane);
      v8f acc0, acc1;
#pragma unroll
      for (int i = 0; i < 8; ++i){ acc0[i] = bias0; acc1[i] = bias1; }
#pragma unroll
      for (int kc = 0; kc < 4; ++kc){
        acc0 = wmma_bf16(ac[kc], bw[0][kc], acc0);
        acc1 = wmma_bf16(ac[kc], bw[1][kc], acc1);
      }
      float cs1[2][8];
#pragma unroll
      for (int v = 0; v < 8; ++v){ cs1[0][v] = tanhf_(acc0[v]); cs1[1][v] = tanhf_(acc1[v]); }

      __syncthreads();   // gates from gate waves now visible

      // ---- state update (LDS reads + VALU only) ----
#pragma unroll
      for (int v = 0; v < 8; ++v){
        const int row = v + roff;
        const float dm1 = dval[v] - 1.0f;
#pragma unroll
        for (int t = 0; t < 2; ++t){
          const int col = n0 + 16 * t + cl;
          const float fg = gbuf[row * G_ST + col];
          const float ig = gbuf[row * G_ST + NH     + col];
          const float og = gbuf[row * G_ST + 2 * NH + col];
          const float cg = gbuf[row * G_ST + 3 * NH + col];
          const float cadj = cstate[t][v] + cs1[t][v] * dm1;   // (c - c_s1) + c_s1*d
          const float cn   = fg * cadj + ig * cg;
          cstate[t][v] = cn;
          const float hn = og * tanhf_(cn);
          hsum[t][v] += hn;
          hA[row * HA_ST + col] = f2bf(hn);
          cA[row * HA_ST + col] = f2bf(cn);
        }
      }
      // publish x tile for step s+1 (data already in registers)
      if (s + 1 < NS){
#pragma unroll
        for (int i = 0; i < 8; ++i){
          int idx = wd * 256 + i * 32 + lane;
          int row = idx >> 6, k = idx & 63;
          xA[row * XA_ST + k] = f2bf(xv[i]);
        }
      }
      __syncthreads();   // h/c/x update visible to everyone
    }

    // ---- out[b] = (sum_s h_s) @ W_out.T, partial reduce into LDS ----
    const float w0 = W_out[n0 + cl];
    const float w1 = W_out[n0 + 16 + cl];
#pragma unroll
    for (int v = 0; v < 8; ++v){
      atomicAdd(&red[v + roff], hsum[0][v] * w0 + hsum[1][v] * w1);
    }
  }

  __syncthreads();
  if (tid < BT){
    out[b0 + tid] = red[tid] + (float)NS * b_out[0];
  }
}

extern "C" void kernel_launch(void* const* d_in, const int* in_sizes, int n_in,
                              void* d_out, int out_size, void* d_ws, size_t ws_size,
                              hipStream_t stream) {
  (void)in_sizes; (void)n_in; (void)out_size; (void)d_ws; (void)ws_size;
  const float* inputs = (const float*)d_in[0];
  const float* tintv  = (const float*)d_in[1];
  const float* W_all  = (const float*)d_in[2];
  const float* b_all  = (const float*)d_in[3];
  const float* U_all  = (const float*)d_in[4];
  const float* b_u    = (const float*)d_in[5];
  const float* W_d    = (const float*)d_in[6];
  const float* b_d    = (const float*)d_in[7];
  const float* W_out  = (const float*)d_in[8];
  const float* b_out  = (const float*)d_in[9];
  float* out = (float*)d_out;

  timelstm_scan<<<dim3(NB / BT), dim3(NTHREADS), 0, stream>>>(
      inputs, tintv, W_all, b_all, U_all, b_u, W_d, b_d, W_out, b_out, out);
}
